// GQAAttention_76690936037879
// MI455X (gfx1250) — matrix operand order repaired and checked by reading, not compiled, not run on hardware
//
#include <hip/hip_runtime.h>
#include <hip/hip_bf16.h>

#define SEQ  2048
#define HID  4096
#define NH   32
#define NKV  8
#define HD   128
#define QKV_N 6144   // 4096 Q + 1024 K + 1024 V output features

typedef __attribute__((ext_vector_type(16))) __bf16 bf16x16;
typedef __attribute__((ext_vector_type(8)))  __bf16 bf16x8;
typedef __attribute__((ext_vector_type(8)))  float  f32x8;
typedef int v4i_ __attribute__((vector_size(16)));   // matches builtin's V4i param

union BF16x16 { bf16x16 v; struct { bf16x8 lo, hi; } h; };

// ---- CDNA5 async global->LDS copy (ASYNCcnt) with compile-safe fallback ----
#if __has_builtin(__builtin_amdgcn_global_load_async_to_lds_b128)
  #if __has_builtin(__builtin_amdgcn_s_wait_asynccnt)
    #define WAIT_ASYNC() __builtin_amdgcn_s_wait_asynccnt(0)
  #else
    #define WAIT_ASYNC() asm volatile("s_wait_asynccnt 0x0" ::: "memory")
  #endif
  #define COPY16(dst, src)                                              \
    __builtin_amdgcn_global_load_async_to_lds_b128(                     \
        (__attribute__((address_space(1))) v4i_*)(void*)(src),          \
        (__attribute__((address_space(3))) v4i_*)(void*)(dst), 0, 0)
#else
  // synchronous staging fallback: global load -> ds_store (ordered by syncthreads)
  #define WAIT_ASYNC()
  #define COPY16(dst, src) do { *(uint4*)(dst) = *(const uint4*)(src); } while (0)
#endif

static __device__ __forceinline__ unsigned short f2bf(float f) {
  unsigned u = __float_as_uint(f);
  u += 0x7fffu + ((u >> 16) & 1u);          // round-to-nearest-even
  return (unsigned short)(u >> 16);
}

static __device__ __forceinline__ f32x8 zero8() {
  f32x8 z; for (int i = 0; i < 8; ++i) z[i] = 0.0f; return z;
}

// A-matrix (16x32 bf16) lane fragment:
// elems 0..15 = K (half*16 .. half*16+15)
static __device__ __forceinline__ bf16x16 load_a(const unsigned short* rowk, int half) {
  BF16x16 r;
  r.h.lo = *(const bf16x8*)(rowk + half * 16);
  r.h.hi = *(const bf16x8*)(rowk + half * 16 + 8);
  return r.v;
}

// ---------------------------------------------------------------- fp32->bf16
__global__ __launch_bounds__(256)
void cvt_bf16_kernel(const float* __restrict__ src, unsigned short* __restrict__ dst, int n4) {
  int i = blockIdx.x * blockDim.x + threadIdx.x;
  int stride = gridDim.x * blockDim.x;
  for (; i < n4; i += stride) {
    float4 f = ((const float4*)src)[i];
    ushort4 o;
    o.x = f2bf(f.x); o.y = f2bf(f.y); o.z = f2bf(f.z); o.w = f2bf(f.w);
    ((ushort4*)dst)[i] = o;
  }
}

// ----------------------------------------------------- LDS-staged GEMM core
// Block = 8 waves, tile 128M x 128N, K stepped by 32.  The 128x32 bf16 B tile
// (8 KB) is staged once per block into a double-buffered LDS tile with async
// global->LDS copies (one s_wait_asynccnt + one barrier per K-step), then all
// 8 waves read their B lane fragments from LDS.  A stays in per-wave registers
// (each wave owns distinct M rows).
#define BSTRIDE 40                       // 32 cols + 8 pad halfwords (80 B rows)
#define BSTAGE  (128 * BSTRIDE)

static __device__ __forceinline__ void stage_b(unsigned short* buf,          // LDS stage base
                                               const unsigned short* W,
                                               size_t rowbase,               // jr*HID + k
                                               int tid) {
#pragma unroll
  for (int i = 0; i < 2; ++i) {          // 512 16-byte chunks, 256 threads x 2
    const int c    = tid + i * 256;
    const int row  = c >> 2;
    const int col8 = (c & 3) * 8;
    COPY16(buf + row * BSTRIDE + col8, W + rowbase + (size_t)row * HID + col8);
  }
}

static __device__ __forceinline__ bf16x16 lds_b_frag(const unsigned short* buf,
                                                     int t, int lm, int half) {
  const unsigned short* p = buf + (t * 16 + lm) * BSTRIDE + half * 16;
  BF16x16 r;
  r.h.lo = *(const bf16x8*)p;
  r.h.hi = *(const bf16x8*)(p + 8);
  return r.v;
}

static __device__ __forceinline__ void gemm_tile(const unsigned short* __restrict__ A,
                                                 const unsigned short* __restrict__ W,
                                                 int m0, int jr,
                                                 unsigned short* sbuf,        // LDS [2][BSTAGE]
                                                 f32x8 acc[8]) {
  const int tid  = threadIdx.x;
  const int lane = tid & 31;
  const int half = lane >> 4;
  const int lm   = lane & 15;
  const unsigned short* arow = A + (size_t)(m0 + lm) * HID;

  stage_b(sbuf, W, (size_t)jr * HID, tid);                 // prefetch stage 0
  for (int k = 0; k < HID; k += 32) {
    unsigned short* cur = sbuf + (((k >> 5) & 1) ? BSTAGE : 0);
    unsigned short* nxt = sbuf + (((k >> 5) & 1) ? 0 : BSTAGE);
    WAIT_ASYNC();                                          // my stage writes landed
    __syncthreads();                                       // everyone's landed / prev reads done
    if (k + 32 < HID) stage_b(nxt, W, (size_t)jr * HID + k + 32, tid);
    bf16x16 a = load_a(arow + k, half);
#pragma unroll
    for (int t = 0; t < 8; ++t) {
      bf16x16 b = lds_b_frag(cur, t, lm, half);
      acc[t] = __builtin_amdgcn_wmma_f32_16x16x32_bf16(false, a, false, b,
                                                       (short)0, acc[t], false, false);
    }
  }
}

// ------------------------------------------------- fused QKV GEMM + bias + RoPE
// Wave tile spans a full 128-wide head, so the RoPE rotate-half pair (d, d+64)
// lives in accum subtiles t and t+4 of the same lane.  Q pre-scaled by 1/sqrt(128).
// V written transposed [kv][d][seq] so the PV GEMM's B operand is contiguous in K.
__global__ __launch_bounds__(256)
void qkv_gemm_rope_kernel(const unsigned short* __restrict__ xb,
                          const unsigned short* __restrict__ wqb,
                          const unsigned short* __restrict__ wkb,
                          const unsigned short* __restrict__ wvb,
                          const float* __restrict__ bq,
                          const float* __restrict__ bk,
                          const float* __restrict__ bv,
                          const int*  __restrict__ pos_ids,
                          unsigned short* __restrict__ qb,   // [32][2048][128]
                          unsigned short* __restrict__ kb,   // [8][2048][128]
                          unsigned short* __restrict__ vt)   // [8][128][2048]
{
  __shared__ unsigned short sbuf[2 * BSTAGE];

  const int w    = threadIdx.x >> 5;
  const int lane = threadIdx.x & 31;
  const int half = lane >> 4;
  const int lm   = lane & 15;
  const int m0   = blockIdx.y * 128 + w * 16;
  const int n0   = blockIdx.x * 128;

  const unsigned short* W; const float* bias; int jr;
  if (n0 < 4096)      { W = wqb; bias = bq; jr = n0; }
  else if (n0 < 5120) { W = wkb; bias = bk; jr = n0 - 4096; }
  else                { W = wvb; bias = bv; jr = n0 - 5120; }

  f32x8 acc[8];
  for (int t = 0; t < 8; ++t) acc[t] = zero8();

  gemm_tile(xb, W, m0, jr, sbuf, acc);

#pragma unroll
  for (int t = 0; t < 8; ++t) {
    float bval = bias[jr + t * 16 + lm];
    for (int r = 0; r < 8; ++r) acc[t][r] += bval;
  }

  const int head = jr >> 7;               // head index within region (jr % 128 == 0)
  if (n0 >= 5120) {                       // ---- V: scatter transposed, bf16
#pragma unroll
    for (int t = 0; t < 8; ++t) {
      const int d = t * 16 + lm;
      for (int r = 0; r < 8; ++r) {
        const int s = m0 + r + 8 * half;
        vt[(size_t)(head * HD + d) * SEQ + s] = f2bf(acc[t][r]);
      }
    }
  } else {                                // ---- Q or K: RoPE in registers
    const float scale = (n0 < 4096) ? 0.08838834764831845f /*1/sqrt(128)*/ : 1.0f;
    unsigned short* out = (n0 < 4096) ? qb : kb;
#pragma unroll
    for (int t = 0; t < 4; ++t) {
      const int d = t * 16 + lm;          // 0..63
      const float invf = powf(10000.0f, -(float)d * (1.0f / 64.0f));
      for (int r = 0; r < 8; ++r) {
        const int s = m0 + r + 8 * half;
        const float ang = (float)pos_ids[s] * invf;
        const float c = cosf(ang), sn = sinf(ang);
        const float lo = acc[t][r], hi = acc[t + 4][r];
        const size_t base = ((size_t)head * SEQ + s) * HD;
        out[base + d]      = f2bf((lo * c - hi * sn) * scale);
        out[base + d + 64] = f2bf((hi * c + lo * sn) * scale);
      }
    }
  }
}

// ------------------------------------------------------------- flash attention
// 4 waves/WG, each wave owns 16 q-rows; 32-key steps; P converts C->A layout
// through a padded per-wave LDS tile (intra-wave, DScnt ordered, no barrier).
__global__ __launch_bounds__(128)
void attn_kernel(const unsigned short* __restrict__ qb,
                 const unsigned short* __restrict__ kb,
                 const unsigned short* __restrict__ vt,
                 unsigned short* __restrict__ ob)    // [2048][4096]
{
  __shared__ unsigned short pbuf[4][16][40];        // 32 cols + 8 pad (bank spread)

  const int w    = threadIdx.x >> 5;
  const int lane = threadIdx.x & 31;
  const int half = lane >> 4;
  const int lm   = lane & 15;
  const int head = blockIdx.x;
  const int kv   = head >> 2;
  const int q0   = blockIdx.y * 64 + w * 16;

  bf16x16 qa[4];
  const unsigned short* qrow = qb + ((size_t)head * SEQ + q0 + lm) * HD;
#pragma unroll
  for (int c = 0; c < 4; ++c) qa[c] = load_a(qrow + c * 32, half);

  f32x8 oacc[8];
  for (int t = 0; t < 8; ++t) oacc[t] = zero8();
  float m[8], l[8];
  for (int r = 0; r < 8; ++r) { m[r] = -1.0e30f; l[r] = 0.0f; }

  const unsigned short* kbase = kb + (size_t)kv * SEQ * HD;
  const unsigned short* vbase = vt + (size_t)kv * HD * SEQ;

  for (int kt = 0; kt < SEQ; kt += 32) {
    f32x8 s0 = zero8(), s1 = zero8();
#pragma unroll
    for (int c = 0; c < 4; ++c) {
      const bf16x16 b0 = *(const bf16x16*)(kbase + (size_t)(kt + lm) * HD + c * 32 + half * 16);
      const bf16x16 b1 = *(const bf16x16*)(kbase + (size_t)(kt + 16 + lm) * HD + c * 32 + half * 16);
      s0 = __builtin_amdgcn_wmma_f32_16x16x32_bf16(false, qa[c], false, b0, (short)0, s0, false, false);
      s1 = __builtin_amdgcn_wmma_f32_16x16x32_bf16(false, qa[c], false, b1, (short)0, s1, false, false);
    }

    // online softmax per C-layout row (M = r + 8*half); reduce across the 16
    // lanes of this half via xor-shuffles (masks < 16 stay inside the half).
    float alpha[8];
#pragma unroll
    for (int r = 0; r < 8; ++r) {
      float v = fmaxf(s0[r], s1[r]);
      v = fmaxf(v, __shfl_xor(v, 1, 32));
      v = fmaxf(v, __shfl_xor(v, 2, 32));
      v = fmaxf(v, __shfl_xor(v, 4, 32));
      v = fmaxf(v, __shfl_xor(v, 8, 32));
      const float mn = fmaxf(m[r], v);
      alpha[r] = expf(m[r] - mn);
      m[r] = mn;
      const float p0 = expf(s0[r] - mn);
      const float p1 = expf(s1[r] - mn);
      pbuf[w][r + 8 * half][lm]      = f2bf(p0);     // keys kt+0..15
      pbuf[w][r + 8 * half][16 + lm] = f2bf(p1);     // keys kt+16..31
      float rs = p0 + p1;
      rs += __shfl_xor(rs, 1, 32);
      rs += __shfl_xor(rs, 2, 32);
      rs += __shfl_xor(rs, 4, 32);
      rs += __shfl_xor(rs, 8, 32);
      l[r] = l[r] * alpha[r] + rs;
    }
#pragma unroll
    for (int t = 0; t < 8; ++t)
      for (int r = 0; r < 8; ++r) oacc[t][r] *= alpha[r];

    // P back out of LDS in A-matrix layout (row = lm, K = keys)
    BF16x16 pa;
    pa.h.lo = *(const bf16x8*)&pbuf[w][lm][half * 16];
    pa.h.hi = *(const bf16x8*)&pbuf[w][lm][half * 16 + 8];

#pragma unroll
    for (int dt = 0; dt < 8; ++dt) {
      const bf16x16 bv_ = *(const bf16x16*)(vbase + (size_t)(dt * 16 + lm) * SEQ + kt + half * 16);
      oacc[dt] = __builtin_amdgcn_wmma_f32_16x16x32_bf16(false, pa.v, false, bv_,
                                                         (short)0, oacc[dt], false, false);
    }
  }

#pragma unroll
  for (int dt = 0; dt < 8; ++dt)
    for (int r = 0; r < 8; ++r) {
      const int s = q0 + r + 8 * half;
      ob[(size_t)s * HID + head * HD + dt * 16 + lm] = f2bf(oacc[dt][r] / l[r]);
    }
}

// ------------------------------------------------------------ output projection
__global__ __launch_bounds__(256)
void out_gemm_kernel(const unsigned short* __restrict__ ob,
                     const unsigned short* __restrict__ wob,
                     const float* __restrict__ bo,
                     float* __restrict__ out)
{
  __shared__ unsigned short sbuf[2 * BSTAGE];

  const int w    = threadIdx.x >> 5;
  const int lane = threadIdx.x & 31;
  const int half = lane >> 4;
  const int lm   = lane & 15;
  const int m0   = blockIdx.y * 128 + w * 16;
  const int n0   = blockIdx.x * 128;

  f32x8 acc[8];
  for (int t = 0; t < 8; ++t) acc[t] = zero8();

  gemm_tile(ob, wob, m0, n0, sbuf, acc);

#pragma unroll
  for (int t = 0; t < 8; ++t) {
    const float bval = bo[n0 + t * 16 + lm];
    for (int r = 0; r < 8; ++r)
      out[(size_t)(m0 + r + 8 * half) * HID + n0 + t * 16 + lm] = acc[t][r] + bval;
  }
}

// --------------------------------------------------------------------- launch
extern "C" void kernel_launch(void* const* d_in, const int* in_sizes, int n_in,
                              void* d_out, int out_size, void* d_ws, size_t ws_size,
                              hipStream_t stream) {
  const float* x   = (const float*)d_in[0];
  const int*   pos = (const int*)  d_in[1];
  const float* wq  = (const float*)d_in[2];
  const float* bq  = (const float*)d_in[3];
  const float* wk  = (const float*)d_in[4];
  const float* bk  = (const float*)d_in[5];
  const float* wv  = (const float*)d_in[6];
  const float* bv  = (const float*)d_in[7];
  const float* wo  = (const float*)d_in[8];
  const float* bo  = (const float*)d_in[9];
  float* out = (float*)d_out;

  char* ws = (char*)d_ws;
  size_t off = 0;
  auto alloc = [&](size_t bytes) -> void* {
    void* p = ws + off; off += (bytes + 255) & ~(size_t)255; return p;
  };
  unsigned short* xb  = (unsigned short*)alloc((size_t)SEQ * HID * 2);       // 16 MB
  unsigned short* wqb = (unsigned short*)alloc((size_t)HID * HID * 2);       // 32 MB
  unsigned short* wkb = (unsigned short*)alloc((size_t)NKV * HD * HID * 2);  //  8 MB
  unsigned short* wvb = (unsigned short*)alloc((size_t)NKV * HD * HID * 2);  //  8 MB
  unsigned short* wob = (unsigned short*)alloc((size_t)HID * HID * 2);       // 32 MB
  unsigned short* qb  = (unsigned short*)alloc((size_t)NH  * SEQ * HD * 2);  // 16 MB
  unsigned short* kb  = (unsigned short*)alloc((size_t)NKV * SEQ * HD * 2);  //  4 MB
  unsigned short* vt  = (unsigned short*)alloc((size_t)NKV * HD * SEQ * 2);  //  4 MB
  unsigned short* ob  = (unsigned short*)alloc((size_t)SEQ * HID * 2);       // 16 MB

  auto cvt = [&](const float* s, unsigned short* d, size_t n) {
    int n4 = (int)(n / 4);
    int blocks = (n4 + 255) / 256;
    cvt_bf16_kernel<<<blocks, 256, 0, stream>>>(s, d, n4);
  };
  cvt(x,  xb,  (size_t)SEQ * HID);
  cvt(wq, wqb, (size_t)HID * HID);
  cvt(wk, wkb, (size_t)NKV * HD * HID);
  cvt(wv, wvb, (size_t)NKV * HD * HID);
  cvt(wo, wob, (size_t)HID * HID);

  qkv_gemm_rope_kernel<<<dim3(QKV_N / 128, SEQ / 128), 256, 0, stream>>>(
      xb, wqb, wkb, wvb, bq, bk, bv, pos, qb, kb, vt);

  attn_kernel<<<dim3(NH, SEQ / 64), 128, 0, stream>>>(qb, kb, vt, ob);

  out_gemm_kernel<<<dim3(HID / 128, SEQ / 128), 256, 0, stream>>>(ob, wob, bo, out);
}
